// ThermodynamicAttentionMechanism_46024869544445
// MI455X (gfx1250) — matrix operand order repaired and checked by reading, not compiled, not run on hardware
//
#include <hip/hip_runtime.h>
#include <hip/hip_bf16.h>

typedef __bf16 bf16_t;
typedef __attribute__((ext_vector_type(16))) __bf16 bf16x16;
typedef __attribute__((ext_vector_type(8)))  float  f32x8;

union F8    { f32x8 v;   float  f[8];  };
union BF16F { bf16x16 v; bf16_t h[16]; uint4 q[2]; };

#define EPSC 1e-8f

// ---------------------------------------------------------------- reductions
__device__ inline float wave_sum(float x) {
#pragma unroll
  for (int off = 16; off > 0; off >>= 1) x += __shfl_xor(x, off, 32);
  return x;
}
__device__ inline float wave_max(float x) {
#pragma unroll
  for (int off = 16; off > 0; off >>= 1) x = fmaxf(x, __shfl_xor(x, off, 32));
  return x;
}

// ---------------------------------------------------------------- fragments
// A-matrix 16x32 bf16 fragment: lane r=lane&15, hb=(lane>>4)*8.
// VGPR0..7 hold K=2*hb..2*hb+15 (contiguous 32B).
__device__ inline bf16x16 ld_afrag_bf16(const bf16_t* row /* row base + kk */, int hb) {
  BF16F f;
  f.q[0] = *(const uint4*)(row + 2 * hb);
  f.q[1] = *(const uint4*)(row + 2 * hb + 8);
  return f.v;
}
// B-matrix 32x16 bf16 fragment: lane n=lane&15, h=lane>>4 holds K=h*16..h*16+15
// contiguous (32B) from an "n-major, K-contiguous" matrix row.
__device__ inline bf16x16 ld_bfrag_bf16(const bf16_t* p /* base + n*ld + kk + h*16 */) {
  BF16F f;
  f.q[0] = *(const uint4*)(p);
  f.q[1] = *(const uint4*)(p + 8);
  return f.v;
}
__device__ inline f32x8 wmma_bf16(bf16x16 a, bf16x16 b, f32x8 c) {
  return __builtin_amdgcn_wmma_f32_16x16x32_bf16(false, a, false, b, (short)0, c,
                                                 false, false);
}

// ---------------------------------------------------------------- kernel 0
// One-shot fp32 -> bf16 conversion (8 elements/thread, 128b in / 128b out)
__global__ __launch_bounds__(256) void cvt_bf16_kernel(const float* __restrict__ src,
                                                       bf16_t* __restrict__ dst,
                                                       int n8) {
  const int i = blockIdx.x * 256 + threadIdx.x;
  if (i >= n8) return;
  const float4* s = (const float4*)src + (size_t)i * 2;
  float4 a = s[0], b = s[1];
  union { uint4 q; bf16_t h[8]; } pk;
  pk.h[0] = (bf16_t)a.x; pk.h[1] = (bf16_t)a.y;
  pk.h[2] = (bf16_t)a.z; pk.h[3] = (bf16_t)a.w;
  pk.h[4] = (bf16_t)b.x; pk.h[5] = (bf16_t)b.y;
  pk.h[6] = (bf16_t)b.z; pk.h[7] = (bf16_t)b.w;
  ((uint4*)dst)[i] = pk.q;
}

// ---------------------------------------------------------------- kernel 1
// C[8192,1024] = X @ W^T + bias with bf16 operands, fp32 accumulate.
// Output bf16 row-major, or transposed [D][S] per batch (for V).
__global__ __launch_bounds__(256) void qkv_gemm(const bf16_t* __restrict__ X,
                                               const bf16_t* __restrict__ W,
                                               const float* __restrict__ bias,
                                               bf16_t* __restrict__ outRM,
                                               bf16_t* __restrict__ outVT,
                                               int transposed) {
  const int rt = blockIdx.x;              // 0..511 : 16-row tile
  const int cg = blockIdx.y;              // 0..1   : 512-col group
  const int w = threadIdx.x >> 5, lane = threadIdx.x & 31;
  const int n = lane & 15, h = lane >> 4, hb = h * 8;
  const int r0 = rt * 16;
  const int c0 = cg * 512 + w * 64;       // this wave: 16 rows x 64 cols

  const bf16_t* xrow = X + (size_t)(r0 + (lane & 15)) * 1024;

  F8 acc[4];
#pragma unroll
  for (int j = 0; j < 4; ++j)
#pragma unroll
    for (int v = 0; v < 8; ++v) acc[j].f[v] = 0.f;

#pragma unroll 2
  for (int kk = 0; kk < 1024; kk += 32) {
    bf16x16 a = ld_afrag_bf16(xrow + kk, hb);
#pragma unroll
    for (int j = 0; j < 4; ++j) {
      const bf16_t* wrow = W + (size_t)(c0 + j * 16 + n) * 1024 + kk + h * 16;
      acc[j].v = wmma_bf16(a, ld_bfrag_bf16(wrow), acc[j].v);
    }
  }

#pragma unroll
  for (int j = 0; j < 4; ++j) {
    const int col = c0 + j * 16 + n;
    const float bv = bias[col];
    if (!transposed) {
#pragma unroll
      for (int v = 0; v < 8; ++v)
        outRM[(size_t)(r0 + h * 8 + v) * 1024 + col] = (bf16_t)(acc[j].f[v] + bv);
    } else {
      // VT[b][d][s]: 8 consecutive s per lane -> one 16B store
      union { uint4 q; bf16_t hh[8]; } pk;
#pragma unroll
      for (int v = 0; v < 8; ++v) pk.hh[v] = (bf16_t)(acc[j].f[v] + bv);
      const int bb = r0 >> 11;
      const int s0 = (r0 & 2047) + h * 8;
      *(uint4*)(outVT + ((size_t)bb * 1024 + col) * 2048 + s0) = pk.q;
    }
  }
}

// ---------------------------------------------------------------- kernel 2
// Fused attention for one (batch, 16-query-row) block. Dynamic LDS:
//   Arow[16][2048] f32 (128KB) | Qs[16][1024] bf16 (32KB) |
//   Ab[16][2048] bf16 (64KB)   | rowscale/rowpmax
__global__ __launch_bounds__(256) void attn_kernel(const bf16_t* __restrict__ Qb,
                                                   const bf16_t* __restrict__ Kb,
                                                   const bf16_t* __restrict__ VT,
                                                   float* __restrict__ A1,
                                                   float* __restrict__ out,
                                                   float* __restrict__ scal,
                                                   const float* __restrict__ tcp,
                                                   const int* __restrict__ tsp) {
  extern __shared__ char smem[];
  float*  Arow = (float*)smem;                                    // 16*2048 f32
  bf16_t* Qs   = (bf16_t*)(smem + 16 * 2048 * 4);                 // 16*1024 bf16
  bf16_t* Ab   = (bf16_t*)(smem + 16 * 2048 * 4 + 16 * 1024 * 2); // 16*2048 bf16
  float*  rsml = (float*)(smem + 16 * 2048 * 4 + 16 * 1024 * 2 + 16 * 2048 * 2);
  float*  rowscale = rsml;       // [16]
  float*  rowpmax  = rsml + 16;  // [16]

  const int b = blockIdx.y;
  const int q0 = blockIdx.x * 16;
  const int tid = threadIdx.x, w = tid >> 5, lane = tid & 31;
  const int n = lane & 15, h = lane >> 4, hb = h * 8;

  const float tc = tcp[0];
  const int ts = tsp[0];
  const float temp = fmaxf(tc * __powf(0.99f, (float)ts), 0.01f);
  const float escale = 1.0f / (32.0f * temp);   // 1/sqrt(1024)/temperature

  { // stage Q tile into LDS (already bf16)
    const uint4* src = (const uint4*)(Qb + ((size_t)b * 2048 + q0) * 1024);
    uint4* dst = (uint4*)Qs;
    for (int i = tid; i < 16 * 1024 / 8; i += 256) dst[i] = src[i];
  }
  __syncthreads();

  // ---- phase 1: energies E = Q K^T * escale into LDS
  for (int ctg = 0; ctg < 4; ++ctg) {
    const int ct0 = w * 16 + ctg * 4;   // this wave's 4 column tiles
    F8 acc[4];
#pragma unroll
    for (int j = 0; j < 4; ++j)
#pragma unroll
      for (int v = 0; v < 8; ++v) acc[j].f[v] = 0.f;

#pragma unroll 2
    for (int kk = 0; kk < 1024; kk += 32) {
      bf16x16 a = ld_afrag_bf16(Qs + (lane & 15) * 1024 + kk, hb);
#pragma unroll
      for (int j = 0; j < 4; ++j) {
        const bf16_t* kr =
            Kb + ((size_t)b * 2048 + (ct0 + j) * 16 + n) * 1024 + kk + h * 16;
        acc[j].v = wmma_bf16(a, ld_bfrag_bf16(kr), acc[j].v);
      }
    }
#pragma unroll
    for (int j = 0; j < 4; ++j)
#pragma unroll
      for (int v = 0; v < 8; ++v)
        Arow[(h * 8 + v) * 2048 + (ct0 + j) * 16 + n] = acc[j].f[v] * escale;
  }
  __syncthreads();

  // ---- phase 2: row softmax + entropy; store A1 to global (heat-flow input)
  for (int rr = 0; rr < 2; ++rr) {
    const int r = w * 2 + rr;
    float* row = Arow + r * 2048;
    float mx = -1e30f;
    for (int c = lane; c < 2048; c += 32) mx = fmaxf(mx, row[c]);
    mx = wave_max(mx);
    float s = 0.f;
    for (int c = lane; c < 2048; c += 32) {
      float g = __expf(row[c] - mx);
      row[c] = g;
      s += g;
    }
    s = wave_sum(s);
    const float inv = 1.0f / s;
    float* arow_g = A1 + ((size_t)b * 2048 + q0 + r) * 2048;
    float H = 0.f;
    for (int c = lane; c < 2048; c += 32) {
      float p = row[c] * inv;
      row[c] = p;
      arow_g[c] = p;
      H -= p * __logf(p + EPSC);
    }
    H = wave_sum(H);
    if (lane == 0) {
      rowscale[r] = __logf(2048.0f) / (H + EPSC);
      rowpmax[r] = inv;   // max(p) = exp(mx-mx)/s = 1/s
    }
  }

  // ---- phase 3: thermal-equilibrium re-softmax + global entropy metrics
  //      final A2 written to bf16 LDS tile for the WMMA A-fragments of phase 4
  float ce = 0.f, sl = 0.f;
  for (int rr = 0; rr < 2; ++rr) {
    const int r = w * 2 + rr;
    float* row = Arow + r * 2048;
    bf16_t* rowb = Ab + r * 2048;
    const float cr = rowscale[r];
    const float m2 = cr * rowpmax[r];
    float s2 = 0.f;
    for (int c = lane; c < 2048; c += 32) {
      float g = __expf(cr * row[c] - m2);
      row[c] = g;
      s2 += g;
    }
    s2 = wave_sum(s2);
    const float inv2 = 1.0f / s2;
    for (int c = lane; c < 2048; c += 32) {
      float p = row[c] * inv2;
      rowb[c] = (bf16_t)p;
      float lp = __logf(p + EPSC);
      ce -= p * lp;
      sl += lp;
    }
  }
  ce = wave_sum(ce);
  sl = wave_sum(sl);
  if (lane == 0) {
    atomicAdd(&scal[0], ce);
    atomicAdd(&scal[1], sl);
  }
  __syncthreads();

  // ---- phase 4: attended = A2 @ V  (A from bf16 LDS, B from transposed V)
  {
    F8 acc[8];
#pragma unroll
    for (int j = 0; j < 8; ++j)
#pragma unroll
      for (int v = 0; v < 8; ++v) acc[j].f[v] = 0.f;

#pragma unroll 2
    for (int kk = 0; kk < 2048; kk += 32) {
      bf16x16 a = ld_afrag_bf16(Ab + (lane & 15) * 2048 + kk, hb);
#pragma unroll
      for (int j = 0; j < 8; ++j) {
        const bf16_t* vr =
            VT + ((size_t)b * 1024 + w * 128 + j * 16 + n) * 2048 + kk + h * 16;
        acc[j].v = wmma_bf16(a, ld_bfrag_bf16(vr), acc[j].v);
      }
    }
#pragma unroll
    for (int j = 0; j < 8; ++j)
#pragma unroll
      for (int v = 0; v < 8; ++v)
        out[((size_t)b * 2048 + q0 + h * 8 + v) * 1024 + w * 128 + j * 16 + n] =
            acc[j].f[v];
  }
}

// ---------------------------------------------------------------- kernel 3
__global__ __launch_bounds__(256) void heatflow_kernel(const float* __restrict__ A1,
                                                       float* __restrict__ scal) {
  const int idx = blockIdx.x * 256 + threadIdx.x;  // S*S elements
  const int q = idx >> 11, k = idx & 2047;
  auto M = [&](int qq) {
    float s = 0.f;
#pragma unroll
    for (int bb = 0; bb < 4; ++bb)
      s += A1[(((size_t)bb * 2048 + qq) << 11) + k];
    return s * 0.25f;
  };
  float g;
  if (q == 0)         g = M(1) - M(0);
  else if (q == 2047) g = M(2047) - M(2046);
  else                g = 0.5f * (M(q + 1) - M(q - 1));
  float v = 0.1f * g;
  float ss = wave_sum(v * v);
  if ((threadIdx.x & 31) == 0) atomicAdd(&scal[2], ss);
}

// ---------------------------------------------------------------- kernel 4
__global__ __launch_bounds__(256) void outmetrics_kernel(const float* __restrict__ out,
                                                         float* __restrict__ scal) {
  const int w = threadIdx.x >> 5, lane = threadIdx.x & 31;
  const int row = blockIdx.x * 8 + w;  // 8192 rows
  const float* p = out + (size_t)row * 1024;
  float ssq = 0.f, mx = -1e30f;
  for (int c = lane; c < 1024; c += 32) {
    float x = p[c];
    ssq += x * x;
    mx = fmaxf(mx, fabsf(x));
  }
  ssq = wave_sum(ssq);
  mx = wave_max(mx);
  float s = 0.f;
  for (int c = lane; c < 1024; c += 32) s += __expf(fabsf(p[c]) - mx);
  s = wave_sum(s);
  const float inv = 1.0f / s;
  float ent = 0.f;
  for (int c = lane; c < 1024; c += 32) {
    float pr = __expf(fabsf(p[c]) - mx) * inv;
    ent -= pr * __logf(pr + EPSC);
  }
  ent = wave_sum(ent);
  if (lane == 0) {
    atomicAdd(&scal[3], sqrtf(ssq));
    atomicAdd(&scal[4], ent);
  }
}

// ---------------------------------------------------------------- kernel 5
__global__ void init_kernel(float* scal) {
  if (threadIdx.x < 8) scal[threadIdx.x] = 0.f;
}

__global__ void finalize_kernel(float* __restrict__ out, const float* __restrict__ scal,
                                const float* __restrict__ tcp,
                                const int* __restrict__ tsp) {
  if (threadIdx.x != 0 || blockIdx.x != 0) return;
  const float tc = tcp[0];
  const int ts = tsp[0];
  const float temp = fmaxf(tc * __powf(0.99f, (float)ts), 0.01f);
  const float ce = scal[0], sl = scal[1], hss = scal[2], nrm = scal[3], oent = scal[4];
  const float S = 2048.f, Bf = 4.f;
  const float ent_prod = __logf(S) * Bf * S - ce;
  const float hf = sqrtf(hss);
  const float u = 1.0f / S;
  const float kl = (u * __logf(u) * Bf * S * S - u * sl) / Bf;
  const float eq = fminf(fmaxf(1.0f - kl, 0.f), 1.f);
  const float fe = nrm / 8192.f - tc * (oent / 8192.f);
  float* t = out + (size_t)8388608;  // after attended [4,2048,1024]
  t[0] = temp;
  t[1] = ent_prod;
  t[2] = hf;
  t[3] = eq;
  t[4] = fe;
}

// ---------------------------------------------------------------- launcher
extern "C" void kernel_launch(void* const* d_in, const int* in_sizes, int n_in,
                              void* d_out, int out_size, void* d_ws, size_t ws_size,
                              hipStream_t stream) {
  const float* query = (const float*)d_in[0];
  const float* key   = (const float*)d_in[1];
  const float* value = (const float*)d_in[2];
  const float* Wq = (const float*)d_in[3];
  const float* bq = (const float*)d_in[4];
  const float* Wk = (const float*)d_in[5];
  const float* bk = (const float*)d_in[6];
  const float* Wv = (const float*)d_in[7];
  const float* bv = (const float*)d_in[8];
  const float* tcp = (const float*)d_in[9];
  const int*   tsp = (const int*)d_in[10];

  char* ws = (char*)d_ws;
  const size_t MB = 1024 * 1024;
  bf16_t* Qb = (bf16_t*)(ws);                  // 16 MiB
  bf16_t* Kb = (bf16_t*)(ws + 16 * MB);        // 16 MiB
  bf16_t* VT = (bf16_t*)(ws + 32 * MB);        // 16 MiB
  float*  A1 = (float*)(ws + 48 * MB);         // 64 MiB [4,2048,2048] f32
  // bf16 copies of the inputs alias the (not-yet-written) A1 region: the
  // projection GEMMs finish before attn_kernel starts writing A1.
  bf16_t* Xq = (bf16_t*)(ws + 48 * MB);        // 16 MiB
  bf16_t* Xk = (bf16_t*)(ws + 64 * MB);        // 16 MiB
  bf16_t* Xv = (bf16_t*)(ws + 80 * MB);        // 16 MiB
  bf16_t* Wbq = (bf16_t*)(ws + 112 * MB);      // 2 MiB each
  bf16_t* Wbk = (bf16_t*)(ws + 114 * MB);
  bf16_t* Wbv = (bf16_t*)(ws + 116 * MB);
  float*  scal = (float*)(ws + 118 * MB);      // 8 floats
  float*  out = (float*)d_out;

  dim3 blk(256);

  init_kernel<<<1, 32, 0, stream>>>(scal);

  // fp32 -> bf16 conversion passes (one-shot, removes cvt from GEMM hot loops)
  const int nX8 = 8192 * 1024 / 8, nW8 = 1024 * 1024 / 8;
  cvt_bf16_kernel<<<dim3(nX8 / 256), blk, 0, stream>>>(query, Xq, nX8);
  cvt_bf16_kernel<<<dim3(nX8 / 256), blk, 0, stream>>>(key, Xk, nX8);
  cvt_bf16_kernel<<<dim3(nX8 / 256), blk, 0, stream>>>(value, Xv, nX8);
  cvt_bf16_kernel<<<dim3(nW8 / 256), blk, 0, stream>>>(Wq, Wbq, nW8);
  cvt_bf16_kernel<<<dim3(nW8 / 256), blk, 0, stream>>>(Wk, Wbk, nW8);
  cvt_bf16_kernel<<<dim3(nW8 / 256), blk, 0, stream>>>(Wv, Wbv, nW8);

  dim3 g1(512, 2);
  qkv_gemm<<<g1, blk, 0, stream>>>(Xq, Wbq, bq, Qb, nullptr, 0);
  qkv_gemm<<<g1, blk, 0, stream>>>(Xk, Wbk, bk, Kb, nullptr, 0);
  qkv_gemm<<<g1, blk, 0, stream>>>(Xv, Wbv, bv, nullptr, VT, 1);

  size_t shmem = 16 * 2048 * 4 + 16 * 1024 * 2 + 16 * 2048 * 2 + 64 * 4;
  (void)hipFuncSetAttribute((const void*)attn_kernel,
                            hipFuncAttributeMaxDynamicSharedMemorySize, (int)shmem);
  attn_kernel<<<dim3(128, 4), blk, shmem, stream>>>(Qb, Kb, VT, A1, out, scal, tcp, tsp);

  heatflow_kernel<<<dim3((2048 * 2048) / 256), blk, 0, stream>>>(A1, scal);
  outmetrics_kernel<<<dim3(1024), blk, 0, stream>>>(out, scal);
  finalize_kernel<<<1, 64, 0, stream>>>(out, scal, tcp, tsp);
}
